// Attention_86672440033867
// MI455X (gfx1250) — compile-verified, matrix operand order checked by reading, not run on hardware
//
#include <hip/hip_runtime.h>
#include <hip/hip_bf16.h>
#include <stddef.h>

// ---------------------------------------------------------------------------
// Types for CDNA5 WMMA (wave32): A/B = v16bf (16x32 / 32x16 bf16), C/D = v8f
// ---------------------------------------------------------------------------
typedef __attribute__((ext_vector_type(16))) __bf16          v16bf;
typedef __attribute__((ext_vector_type(8)))  float           v8f;
typedef __attribute__((ext_vector_type(8)))  unsigned short  u16x8;

#define WMMA_BF16(A, B, C) \
    __builtin_amdgcn_wmma_f32_16x16x32_bf16(false, (A), false, (B), (short)0, (C), false, false)

// Problem constants
#define NTOK 1024           // 32*32 tokens
#define HEADS 8
#define DHEAD 32
#define INNER 256           // HEADS*DHEAD
#define BATCH 16
#define INP 384
#define OUP 384
#define MTOT (BATCH * NTOK) // 16384
#define SCALE 0.17677669529663687f     // 1/sqrt(32)
#define LOG2E 1.4426950408889634f
#define BIAS_HSTRIDE (64 * 64 * 32 * 8)  // per-head swizzled bias elements (1M)

__device__ __forceinline__ unsigned short f2bf(float f) {
    unsigned u = __builtin_bit_cast(unsigned, f);
    u = (u + 0x7FFFu + ((u >> 16) & 1u)) >> 16;   // round-to-nearest-even
    return (unsigned short)u;
}
__device__ __forceinline__ float bf2f(unsigned short s) {
    unsigned u = ((unsigned)s) << 16;
    return __builtin_bit_cast(float, u);
}

// Load a 16x32 bf16 A-layout fragment (also used for B by feeding B^T):
// lane L: halves h=0..7  -> K = k0 + (L>=16?8:0) + h        (contiguous b128)
//         halves h=8..15 -> K = k0 + 16 + (L>=16?8:0) + h-8 (contiguous b128)
__device__ __forceinline__ v16bf load_frag(const unsigned short* p, int stride,
                                           int row, int k0, int lane) {
    const unsigned short* q = p + (size_t)row * stride + k0 + ((lane >> 4) << 3);
    union { v16bf bf; u16x8 u[2]; } r;
    r.u[0] = *(const u16x8*)(q);
    r.u[1] = *(const u16x8*)(q + 16);
    return r.bf;
}

// ---------------------------------------------------------------------------
// Prep kernels
// ---------------------------------------------------------------------------
__global__ __launch_bounds__(256) void k_convert_x(const float* __restrict__ x,
                                                   unsigned short* __restrict__ xb) {
    int i = (blockIdx.x * 256 + threadIdx.x) * 4;   // MTOT*INP / 4 threads
    float4 v = *(const float4*)(x + i);
    xb[i + 0] = f2bf(v.x);
    xb[i + 1] = f2bf(v.y);
    xb[i + 2] = f2bf(v.z);
    xb[i + 3] = f2bf(v.w);
}

__global__ __launch_bounds__(256) void k_transpose_wqkv(const float* __restrict__ w,
                                                        unsigned short* __restrict__ wt) {
    int i = blockIdx.x * 256 + threadIdx.x;         // over 768*384
    int n = i / INP, k = i % INP;
    wt[n * INP + k] = f2bf(w[k * (3 * INNER) + n]);
}

__global__ __launch_bounds__(256) void k_transpose_wout(const float* __restrict__ w,
                                                        unsigned short* __restrict__ wt) {
    int i = blockIdx.x * 256 + threadIdx.x;         // over 384*256
    int n = i / INNER, k = i % INNER;
    wt[n * INNER + k] = f2bf(w[k * OUP + n]);
}

// Swizzled bias: bsw[h][qt][kt][lane][v] = log2e * table[rel_index[n,m]][h]
// where n = qt*16 + (lane>>4)*8 + v, m = kt*16 + (lane&15)  -> exactly the
// element (lane, v) of the 16x16 WMMA C/D tile (qt, kt). One b128 per tile.
__global__ __launch_bounds__(256) void k_bias_gather(const int* __restrict__ ridx,
                                                     const float* __restrict__ table,
                                                     unsigned short* __restrict__ bsw) {
    int id = blockIdx.x * 256 + threadIdx.x;        // over 1024*1024 (n*1024+m)
    int n = id >> 10, m = id & 1023;
    int idx = ridx[id];
    const float* t = table + (size_t)idx * HEADS;
    int qt = n >> 4, r = n & 15;
    int kt = m >> 4, c = m & 15;
    int lane = ((r >> 3) << 4) | c;                 // (hg, rl)
    int v = r & 7;
    size_t base = ((((size_t)qt * 64 + kt) * 32 + lane) * 8 + v);
#pragma unroll
    for (int h = 0; h < HEADS; ++h)
        bsw[(size_t)h * BIAS_HSTRIDE + base] = f2bf(t[h] * LOG2E);
}

// ---------------------------------------------------------------------------
// GEMM1: qkv = x_bf[16384,384] @ Wqkv  (B fed as B^T[768,384] bf16)
// Epilogue splits into Q (scaled by SCALE*log2e, [b,h,n,d]), K ([b,h,n,d]),
// V^T ([b,h,d,n]).  8 waves, block tile 128x128, wave tile 32x64.
// Routing is tile-uniform (readfirstlane) -> scalar branches; V tiles store
// straight from registers as contiguous b128 (8 consecutive tokens per lane);
// Q/K tiles transpose through 512B/wave LDS -> coalesced b128 stores.
// ---------------------------------------------------------------------------
__global__ __launch_bounds__(256) void k_gemm_qkv(const unsigned short* __restrict__ A,
                                                  const unsigned short* __restrict__ Bt,
                                                  unsigned short* __restrict__ qw,
                                                  unsigned short* __restrict__ kw,
                                                  unsigned short* __restrict__ vtw) {
    __shared__ unsigned short sT[8 * 256];          // 16x16 bf16 tile per wave

    const int lane = threadIdx.x & 31, wave = threadIdx.x >> 5;
    const int rl = lane & 15, hg = lane >> 4;
    const int wm = wave & 3, wn = wave >> 2;
    const int Mbase = blockIdx.y * 128 + wm * 32;
    const int Nbase = blockIdx.x * 128 + wn * 64;
    unsigned short* myT = sT + wave * 256;

    v8f acc[2][4];
#pragma unroll
    for (int i = 0; i < 2; ++i)
#pragma unroll
        for (int j = 0; j < 4; ++j) acc[i][j] = (v8f){};

    for (int ks = 0; ks < INP; ks += 32) {
        v16bf a0 = load_frag(A, INP, Mbase + rl, ks, lane);
        v16bf a1 = load_frag(A, INP, Mbase + 16 + rl, ks, lane);
        v16bf b[4];
#pragma unroll
        for (int j = 0; j < 4; ++j)
            b[j] = load_frag(Bt, INP, Nbase + j * 16 + rl, ks, lane);
#pragma unroll
        for (int j = 0; j < 4; ++j) {
            acc[0][j] = WMMA_BF16(a0, b[j], acc[0][j]);
            acc[1][j] = WMMA_BF16(a1, b[j], acc[1][j]);
        }
    }

#pragma unroll
    for (int nt = 0; nt < 4; ++nt) {
        // Tile-uniform routing: force into SGPRs for scalar branches.
        int colbase = __builtin_amdgcn_readfirstlane(Nbase + nt * 16);
        int part = colbase >> 8;                    // 0=q 1=k 2=v
        int h    = (colbase >> 5) & 7;
        int d0   = colbase & 31;                    // 0 or 16
#pragma unroll
        for (int mt = 0; mt < 2; ++mt) {
            int rowb = Mbase + mt * 16;             // tile token base
            if (part == 2) {
                // V^T[d][n]: lane holds 8 consecutive tokens at fixed d.
                int tok0 = rowb + hg * 8;
                int b = tok0 >> 10, n0 = tok0 & 1023;
                int d = d0 + rl;
                union { u16x8 u; unsigned short s[8]; } pk;
#pragma unroll
                for (int v = 0; v < 8; ++v) pk.s[v] = f2bf(acc[mt][nt][v]);
                *(u16x8*)(vtw + ((size_t)(b * HEADS + h) * DHEAD + d) * NTOK + n0) = pk.u;
            } else {
                // Q/K[n][d]: transpose 16x16 tile through wave-private LDS.
                float sc = (part == 0) ? (SCALE * LOG2E) : 1.0f;
#pragma unroll
                for (int v = 0; v < 8; ++v)
                    myT[(hg * 8 + v) * 16 + rl] = f2bf(acc[mt][nt][v] * sc);
                __builtin_amdgcn_wave_barrier();
                u16x8 rowv = *(const u16x8*)(myT + rl * 16 + hg * 8);
                int tok = rowb + rl;
                int b = tok >> 10, n = tok & 1023;
                unsigned short* dst = (part == 0) ? qw : kw;
                *(u16x8*)(dst + ((size_t)(b * HEADS + h) * NTOK + n) * DHEAD + d0 + hg * 8) = rowv;
                __builtin_amdgcn_wave_barrier();
            }
        }
    }
}

// ---------------------------------------------------------------------------
// Fused flash attention: grid (N/128, HEADS, BATCH), block 256 (8 waves).
// Wave owns 16 query rows; loop over 1024 keys in blocks of 64.
// Scores are already in log2-domain (Q pre-scaled, bias pre-scaled) -> exp2.
// ---------------------------------------------------------------------------
__global__ __launch_bounds__(256) void k_attention(const unsigned short* __restrict__ qw,
                                                   const unsigned short* __restrict__ kw,
                                                   const unsigned short* __restrict__ vtw,
                                                   const unsigned short* __restrict__ bsw,
                                                   unsigned short* __restrict__ aout) {
    __shared__ unsigned short sP[8 * 16 * 64];      // 2KB per wave, wave-private

    const int lane = threadIdx.x & 31, wave = threadIdx.x >> 5;
    const int rl = lane & 15, hg = lane >> 4;
    const int b = blockIdx.z, h = blockIdx.y;
    const int qrow0 = blockIdx.x * 128 + wave * 16;
    const int qt = qrow0 >> 4;

    const size_t bh = (size_t)(b * HEADS + h);
    const unsigned short* Q  = qw  + bh * NTOK * DHEAD;
    const unsigned short* K  = kw  + bh * NTOK * DHEAD;
    const unsigned short* Vt = vtw + bh * DHEAD * NTOK;
    const unsigned short* Bq = bsw + (size_t)h * BIAS_HSTRIDE
                                   + ((size_t)qt * 64) * 256 + lane * 8;
    unsigned short* myP = sP + wave * 1024;

    v16bf aq = load_frag(Q, DHEAD, qrow0 + rl, 0, lane);

    float m[8], lsum[8];
#pragma unroll
    for (int v = 0; v < 8; ++v) { m[v] = -3.0e38f; lsum[v] = 0.0f; }
    v8f o0 = {}, o1 = {};

    for (int kb = 0; kb < NTOK; kb += 64) {
        const int kt0 = kb >> 4;
        // ---- S = (Q*scale*log2e) K^T  (4 x 16x16 tiles) ----
        v8f s[4];
#pragma unroll
        for (int j = 0; j < 4; ++j) {
            v16bf kf = load_frag(K, DHEAD, kb + j * 16 + rl, 0, lane);
            v8f z = {};
            s[j] = WMMA_BF16(aq, kf, z);
        }
        // ---- + relative position bias (fragment-swizzled, one b128/tile) ----
#pragma unroll
        for (int j = 0; j < 4; ++j) {
            u16x8 bb = *(const u16x8*)(Bq + (size_t)(kt0 + j) * 256);
#pragma unroll
            for (int v = 0; v < 8; ++v) s[j][v] += bf2f(bb[v]);
        }

        // ---- online softmax in log2 domain (rows live in 16-lane groups) ----
#pragma unroll
        for (int v = 0; v < 8; ++v) {
            float mx = fmaxf(fmaxf(s[0][v], s[1][v]), fmaxf(s[2][v], s[3][v]));
#pragma unroll
            for (int d = 1; d < 16; d <<= 1) mx = fmaxf(mx, __shfl_xor(mx, d));
            float mn = fmaxf(m[v], mx);
            float alpha = __builtin_amdgcn_exp2f(m[v] - mn);
            m[v] = mn;
            float rs = 0.0f;
#pragma unroll
            for (int j = 0; j < 4; ++j) {
                float p = __builtin_amdgcn_exp2f(s[j][v] - mn);
                s[j][v] = p;
                rs += p;
            }
#pragma unroll
            for (int d = 1; d < 16; d <<= 1) rs += __shfl_xor(rs, d);
            lsum[v] = lsum[v] * alpha + rs;
            o0[v] *= alpha;
            o1[v] *= alpha;
        }

        // ---- D-layout P -> LDS -> A-layout P (bf16); wave-private region,
        //      LDS pipeline is in-order per wave, so no workgroup barrier ----
#pragma unroll
        for (int j = 0; j < 4; ++j)
#pragma unroll
            for (int v = 0; v < 8; ++v)
                myP[(hg * 8 + v) * 64 + j * 16 + rl] = f2bf(s[j][v]);
        __builtin_amdgcn_wave_barrier();
        union { v16bf bf; u16x8 u[2]; } p0, p1;
        const unsigned short* pr = myP + rl * 64 + hg * 8;
        p0.u[0] = *(const u16x8*)(pr);
        p0.u[1] = *(const u16x8*)(pr + 16);
        p1.u[0] = *(const u16x8*)(pr + 32);
        p1.u[1] = *(const u16x8*)(pr + 48);
        __builtin_amdgcn_wave_barrier();

        // ---- O += P V   (V^T rows are contiguous in K -> b128 frag loads) ----
        v16bf vf00 = load_frag(Vt, NTOK, rl,      kb, lane);
        v16bf vf01 = load_frag(Vt, NTOK, rl,      kb + 32, lane);
        v16bf vf10 = load_frag(Vt, NTOK, 16 + rl, kb, lane);
        v16bf vf11 = load_frag(Vt, NTOK, 16 + rl, kb + 32, lane);
        o0 = WMMA_BF16(p0.bf, vf00, o0);
        o0 = WMMA_BF16(p1.bf, vf01, o0);
        o1 = WMMA_BF16(p0.bf, vf10, o1);
        o1 = WMMA_BF16(p1.bf, vf11, o1);
    }

    // ---- normalize, transpose through LDS, coalesced b128 stores ----
#pragma unroll
    for (int v = 0; v < 8; ++v) {
        float inv = 1.0f / lsum[v];
        myP[(hg * 8 + v) * 32 + rl]      = f2bf(o0[v] * inv);
        myP[(hg * 8 + v) * 32 + 16 + rl] = f2bf(o1[v] * inv);
    }
    __builtin_amdgcn_wave_barrier();
    {
        int row = rl;                   // lanes 0..15 / 16..31 split the halves
        int half = hg;
        u16x8 t0 = *(const u16x8*)(myP + row * 32 + half * 16);
        u16x8 t1 = *(const u16x8*)(myP + row * 32 + half * 16 + 8);
        size_t base = ((size_t)(b * NTOK + qrow0 + row)) * INNER + h * DHEAD + half * 16;
        *(u16x8*)(aout + base)     = t0;
        *(u16x8*)(aout + base + 8) = t1;
    }
}

// ---------------------------------------------------------------------------
// GEMM2: out = attn[16384,256] @ Wout (fed as Wout^T[384,256]) + b_out  (f32)
// ---------------------------------------------------------------------------
__global__ __launch_bounds__(256) void k_gemm_out(const unsigned short* __restrict__ A,
                                                  const unsigned short* __restrict__ Bt,
                                                  const float* __restrict__ bout,
                                                  float* __restrict__ out) {
    const int lane = threadIdx.x & 31, wave = threadIdx.x >> 5;
    const int rl = lane & 15, hg = lane >> 4;
    const int wm = wave & 3, wn = wave >> 2;
    const int Mbase = blockIdx.y * 128 + wm * 32;
    const int Nbase = blockIdx.x * 128 + wn * 64;

    v8f acc[2][4];
#pragma unroll
    for (int i = 0; i < 2; ++i)
#pragma unroll
        for (int j = 0; j < 4; ++j) acc[i][j] = (v8f){};

    for (int ks = 0; ks < INNER; ks += 32) {
        v16bf a0 = load_frag(A, INNER, Mbase + rl, ks, lane);
        v16bf a1 = load_frag(A, INNER, Mbase + 16 + rl, ks, lane);
        v16bf b[4];
#pragma unroll
        for (int j = 0; j < 4; ++j)
            b[j] = load_frag(Bt, INNER, Nbase + j * 16 + rl, ks, lane);
#pragma unroll
        for (int j = 0; j < 4; ++j) {
            acc[0][j] = WMMA_BF16(a0, b[j], acc[0][j]);
            acc[1][j] = WMMA_BF16(a1, b[j], acc[1][j]);
        }
    }

#pragma unroll
    for (int nt = 0; nt < 4; ++nt) {
        int col = Nbase + nt * 16 + rl;
        float bias = bout[col];
#pragma unroll
        for (int mt = 0; mt < 2; ++mt) {
            int rowb = Mbase + mt * 16 + hg * 8;
#pragma unroll
            for (int v = 0; v < 8; ++v)
                out[(size_t)(rowb + v) * OUP + col] = acc[mt][nt][v] + bias;
        }
    }
}

// ---------------------------------------------------------------------------
// Host launcher
// ---------------------------------------------------------------------------
extern "C" void kernel_launch(void* const* d_in, const int* in_sizes, int n_in,
                              void* d_out, int out_size, void* d_ws, size_t ws_size,
                              hipStream_t stream) {
    (void)in_sizes; (void)n_in; (void)out_size;

    const float* x      = (const float*)d_in[0];
    const float* w_qkv  = (const float*)d_in[1];
    const float* rtable = (const float*)d_in[2];
    const float* w_out  = (const float*)d_in[3];
    const float* b_out  = (const float*)d_in[4];
    const int*   ridx   = (const int*)d_in[5];
    float* out = (float*)d_out;

    // Workspace layout (bytes), 256B-aligned chunks, ~61 MB total
    char* ws = (char*)d_ws;
    size_t off = 0;
    auto alloc = [&](size_t bytes) { char* p = ws + off; off += (bytes + 255) & ~(size_t)255; return p; };
    unsigned short* xb    = (unsigned short*)alloc((size_t)MTOT * INP * 2);
    unsigned short* wqkvT = (unsigned short*)alloc((size_t)(3 * INNER) * INP * 2);
    unsigned short* woutT = (unsigned short*)alloc((size_t)OUP * INNER * 2);
    unsigned short* qw    = (unsigned short*)alloc((size_t)BATCH * HEADS * NTOK * DHEAD * 2);
    unsigned short* kw    = (unsigned short*)alloc((size_t)BATCH * HEADS * NTOK * DHEAD * 2);
    unsigned short* vtw   = (unsigned short*)alloc((size_t)BATCH * HEADS * DHEAD * NTOK * 2);
    unsigned short* biasb = (unsigned short*)alloc((size_t)HEADS * BIAS_HSTRIDE * 2);
    unsigned short* aout  = (unsigned short*)alloc((size_t)MTOT * INNER * 2);
    if (off > ws_size) return;  // workspace too small; bail deterministically

    // 1) prep
    k_convert_x<<<(MTOT * INP) / (256 * 4), 256, 0, stream>>>(x, xb);
    k_transpose_wqkv<<<(3 * INNER * INP) / 256, 256, 0, stream>>>(w_qkv, wqkvT);
    k_transpose_wout<<<(OUP * INNER) / 256, 256, 0, stream>>>(w_out, woutT);
    k_bias_gather<<<(NTOK * NTOK) / 256, 256, 0, stream>>>(ridx, rtable, biasb);

    // 2) QKV projection
    k_gemm_qkv<<<dim3((3 * INNER) / 128, MTOT / 128), 256, 0, stream>>>(xb, wqkvT, qw, kw, vtw);

    // 3) fused attention
    k_attention<<<dim3(NTOK / 128, HEADS, BATCH), 256, 0, stream>>>(qw, kw, vtw, biasb, aout);

    // 4) output projection (+ bias)
    k_gemm_out<<<dim3(OUP / 128, MTOT / 128), 256, 0, stream>>>(aout, woutT, b_out, out);
}
